// MADE_46617575030817
// MI455X (gfx1250) — compile-verified, matrix operand order checked by reading, not run on hardware
//
#include <hip/hip_runtime.h>
#include <hip/hip_bf16.h>
#include <stdint.h>

// Problem constants (match the reference)
#define IN_CH  794
#define OUT_CH 512
#define Z1     1024
#define BATCH  32768
#define K1PAD  832   // IN_CH padded up to a multiple of 32 for the weight buffer

typedef __attribute__((ext_vector_type(16))) __bf16 bf16x16;
typedef __attribute__((ext_vector_type(8)))  float  floatx8;

union FragU { bf16x16 v; uint4 q[2]; };

#define AS1 __attribute__((address_space(1)))
#define AS3 __attribute__((address_space(3)))

// Pointee type must match the builtin's declared parameter exactly:
// "__attribute__((__vector_size__(4 * sizeof(int)))) int *"
typedef int vec4i __attribute__((__vector_size__(4 * sizeof(int))));

#if defined(__AMDGCN__) && __has_builtin(__builtin_amdgcn_global_load_async_to_lds_b128)
#define HAVE_ASYNC_LDS 1
#else
#define HAVE_ASYNC_LDS 0
#endif

__device__ __forceinline__ void wait_async0() {
#if __has_builtin(__builtin_amdgcn_s_wait_asynccnt)
  __builtin_amdgcn_s_wait_asynccnt(0);
#elif defined(__AMDGCN__)
  asm volatile("s_wait_asynccnt 0x0" ::: "memory");
#endif
}

// fp32 -> bf16 (RNE). Prefer the packed HW convert when the builtin exists.
__device__ __forceinline__ unsigned short f32_to_bf16(float f) {
  unsigned int u = __float_as_uint(f);
  u += 0x7FFFu + ((u >> 16) & 1u);
  return (unsigned short)(u >> 16);
}

#if __has_builtin(__builtin_amdgcn_cvt_pk_bf16_f32)
__device__ __forceinline__ unsigned int pack2(float a, float b) {
  typedef __attribute__((ext_vector_type(2))) __bf16 bf16x2_t;
  union { bf16x2_t h; unsigned int u; } c;
  c.h = __builtin_amdgcn_cvt_pk_bf16_f32(a, b);
  return c.u;
}
__device__ __forceinline__ unsigned short bf16_one(float v) {
  return (unsigned short)(pack2(v, 0.0f) & 0xFFFFu);
}
#else
__device__ __forceinline__ unsigned int pack2(float a, float b) {
  return (unsigned int)f32_to_bf16(a) | ((unsigned int)f32_to_bf16(b) << 16);
}
__device__ __forceinline__ unsigned short bf16_one(float v) { return f32_to_bf16(v); }
#endif

// ---------------------------------------------------------------------------
// Weight prep: wbf[z][k] = bf16(W[z][k] * M[z][k])  (zero-padded to Kpad)
//              biasOut[z] = b[z] + bc[z] + sum_k Wc[z][k]*M[z][k]
// One block (256 threads) per output row z.
// ---------------------------------------------------------------------------
__global__ __launch_bounds__(256)
void prep_weights(const float* __restrict__ W, const float* __restrict__ Wc,
                  const float* __restrict__ Msk,
                  const float* __restrict__ b, const float* __restrict__ bc,
                  int K, int Kpad,
                  unsigned short* __restrict__ wbf, float* __restrict__ biasOut) {
  const int z = blockIdx.x;
  const int t = threadIdx.x;
  const size_t ro = (size_t)z * K;
  float s = 0.0f;
  for (int k = t; k < Kpad; k += 256) {
    float wv = 0.0f;
    if (k < K) {
      const float mv = Msk[ro + k];
      wv = W[ro + k] * mv;
      s += Wc[ro + k] * mv;
    }
    wbf[(size_t)z * Kpad + k] = bf16_one(wv);
  }
  __shared__ float red[256];
  red[t] = s;
  __syncthreads();
  #pragma unroll
  for (int off = 128; off > 0; off >>= 1) {
    if (t < off) red[t] += red[t + off];
    __syncthreads();
  }
  if (t == 0) biasOut[z] = b[z] + bc[z] + red[0];
}

// ---------------------------------------------------------------------------
// Tiled WMMA GEMM:  Out[M][N] = relu(A[M][K] * Bw[N][K]^T + bias[N])
//   A: fp32 (converted on the fly, via VGPR staging) or bf16 (async->LDS).
//   B: bf16 weights, staged with GLOBAL_LOAD_ASYNC_TO_LDS_B128 when available.
//   Block: 256 threads = 8 waves, 128x128 output tile.
//   Wave grid 4(M) x 2(N): each wave -> 32x64 = 2x4 WMMA 16x16 tiles.
//   K-loop: 32-wide slices, double-buffered LDS (40 KB of 320 KB WGP LDS).
// ---------------------------------------------------------------------------
#define LDS_STRIDE 40                       // halfwords per tile row (32 data + 8 pad)
#define BUF_HW     (128 * LDS_STRIDE)       // halfwords per A (or B) tile

template<bool A_IS_F32, bool OUT_BF16>
__global__ __launch_bounds__(256)
void gemm_bias_relu(const void* __restrict__ Av,
                    const unsigned short* __restrict__ Bw,
                    const float* __restrict__ bias,
                    void* __restrict__ Outv,
                    int M, int N, int K, int lda, int ldb, int ldo) {
  __shared__ unsigned short smem[2 * 2 * BUF_HW];   // 2 buffers x (A tile + B tile)

  const int tid   = threadIdx.x;
  const int lane  = tid & 31;
  const int wid   = tid >> 5;
  const int waveM = wid & 3;      // 0..3 -> 32-row strip
  const int waveN = wid >> 2;     // 0..1 -> 64-col strip
  const int l15   = lane & 15;
  const int kg    = (lane >> 4) * 8;   // WMMA 16-bit operand K-chunk base

  const int nBase = blockIdx.x * 128;
  const int mBase = blockIdx.y * 128;

  floatx8 acc[2][4] = {};

  const int nSlices = (K + 31) >> 5;
  const int c0 = tid, c1 = tid + 256;  // each thread stages 2 chunks per tile

  // --- VGPR staging helpers (chunk = 8 halfwords = 16B; 4 chunks / 32-wide row) ---
  auto loadA = [&](int s, int c) -> uint4 {
    const int row = c >> 2;
    const int kc  = (c & 3) << 3;
    const int gk  = (s << 5) + kc;
    uint4 r;
    if constexpr (A_IS_F32) {
      const float* ap = (const float*)Av + (size_t)(mBase + row) * lda + gk;
      unsigned int w[4];
      #pragma unroll
      for (int j = 0; j < 4; ++j) {
        float2 p = make_float2(0.0f, 0.0f);
        if (gk + 2 * j < K) p = *(const float2*)(ap + 2 * j);  // K even -> pair safe
        w[j] = pack2(p.x, p.y);
      }
      r.x = w[0]; r.y = w[1]; r.z = w[2]; r.w = w[3];
    } else {
      r = *(const uint4*)((const unsigned short*)Av + (size_t)(mBase + row) * lda + gk);
    }
    return r;
  };
  auto loadB = [&](int s, int c) -> uint4 {
    const int row = c >> 2;
    const int kc  = (c & 3) << 3;
    return *(const uint4*)(Bw + (size_t)(nBase + row) * ldb + (s << 5) + kc);
  };
  auto storeLDS = [&](int buf, int which, int c, uint4 r) {
    const int row = c >> 2;
    const int kc  = (c & 3) << 3;
    unsigned short* p = smem + buf * (2 * BUF_HW) + which * BUF_HW + row * LDS_STRIDE + kc;
    *(uint4*)p = r;
  };

  // --- async global->LDS staging (B tile always; A tile when already bf16) ---
  auto asyncStage = [&](int s, int buf) {
#if HAVE_ASYNC_LDS
    #pragma unroll
    for (int cc = 0; cc < 2; ++cc) {
      const int c   = cc ? c1 : c0;
      const int row = c >> 2;
      const int kc  = (c & 3) << 3;
      {
        const unsigned short* g = Bw + (size_t)(nBase + row) * ldb + (s << 5) + kc;
        unsigned short* l = smem + buf * (2 * BUF_HW) + BUF_HW + row * LDS_STRIDE + kc;
        __builtin_amdgcn_global_load_async_to_lds_b128((AS1 vec4i*)g, (AS3 vec4i*)l, 0, 0);
      }
      if constexpr (!A_IS_F32) {
        const unsigned short* g =
            (const unsigned short*)Av + (size_t)(mBase + row) * lda + (s << 5) + kc;
        unsigned short* l = smem + buf * (2 * BUF_HW) + row * LDS_STRIDE + kc;
        __builtin_amdgcn_global_load_async_to_lds_b128((AS1 vec4i*)g, (AS3 vec4i*)l, 0, 0);
      }
    }
#endif
  };

  auto compute = [&](int buf) {
    const unsigned short* As = smem + buf * (2 * BUF_HW);
    const unsigned short* Bs = As + BUF_HW;
    FragU a[2], b[4];
    #pragma unroll
    for (int mi = 0; mi < 2; ++mi) {
      const unsigned short* p = As + (waveM * 32 + mi * 16 + l15) * LDS_STRIDE + kg;
      a[mi].q[0] = *(const uint4*)p;
      a[mi].q[1] = *(const uint4*)(p + 16);
    }
    #pragma unroll
    for (int ni = 0; ni < 4; ++ni) {
      const unsigned short* p = Bs + (waveN * 64 + ni * 16 + l15) * LDS_STRIDE + kg;
      b[ni].q[0] = *(const uint4*)p;
      b[ni].q[1] = *(const uint4*)(p + 16);
    }
    #pragma unroll
    for (int mi = 0; mi < 2; ++mi)
      #pragma unroll
      for (int ni = 0; ni < 4; ++ni)
        acc[mi][ni] = __builtin_amdgcn_wmma_f32_16x16x32_bf16(
            false, a[mi].v, false, b[ni].v, (short)0, acc[mi][ni], false, false);
  };

  // --- prologue: stage slice 0 into buffer 0 ---
  {
#if HAVE_ASYNC_LDS
    if constexpr (A_IS_F32) {
      uint4 ra0 = loadA(0, c0), ra1 = loadA(0, c1);
      storeLDS(0, 0, c0, ra0); storeLDS(0, 0, c1, ra1);
    }
    asyncStage(0, 0);
    wait_async0();
#else
    uint4 ra0 = loadA(0, c0), ra1 = loadA(0, c1);
    uint4 rb0 = loadB(0, c0), rb1 = loadB(0, c1);
    storeLDS(0, 0, c0, ra0); storeLDS(0, 0, c1, ra1);
    storeLDS(0, 1, c0, rb0); storeLDS(0, 1, c1, rb1);
#endif
  }
  __syncthreads();

  // --- main loop: double-buffered; async copies overlap the WMMA work ---
  for (int s = 0; s < nSlices; ++s) {
    const int buf = s & 1;
    const bool more = (s + 1) < nSlices;
    uint4 ra0 = {}, ra1 = {};
#if !HAVE_ASYNC_LDS
    uint4 rb0 = {}, rb1 = {};
#endif
    if (more) {
#if HAVE_ASYNC_LDS
      if constexpr (A_IS_F32) { ra0 = loadA(s + 1, c0); ra1 = loadA(s + 1, c1); }
      asyncStage(s + 1, buf ^ 1);
#else
      ra0 = loadA(s + 1, c0); ra1 = loadA(s + 1, c1);
      rb0 = loadB(s + 1, c0); rb1 = loadB(s + 1, c1);
#endif
    }
    compute(buf);
    if (more) {
#if HAVE_ASYNC_LDS
      if constexpr (A_IS_F32) {
        storeLDS(buf ^ 1, 0, c0, ra0); storeLDS(buf ^ 1, 0, c1, ra1);
      }
      wait_async0();
#else
      storeLDS(buf ^ 1, 0, c0, ra0); storeLDS(buf ^ 1, 0, c1, ra1);
      storeLDS(buf ^ 1, 1, c0, rb0); storeLDS(buf ^ 1, 1, c1, rb1);
#endif
    }
    __syncthreads();
  }

  // --- epilogue: bias + relu, store per documented C layout ---
  #pragma unroll
  for (int ni = 0; ni < 4; ++ni) {
    const int gc = nBase + waveN * 64 + ni * 16 + l15;
    const float bv = bias[gc];
    #pragma unroll
    for (int mi = 0; mi < 2; ++mi) {
      const int rb = mBase + waveM * 32 + mi * 16 + ((lane >> 4) << 3);
      #pragma unroll
      for (int i = 0; i < 8; ++i) {
        float v = acc[mi][ni][i] + bv;
        v = fmaxf(v, 0.0f);
        if constexpr (OUT_BF16)
          ((unsigned short*)Outv)[(size_t)(rb + i) * ldo + gc] = bf16_one(v);
        else
          ((float*)Outv)[(size_t)(rb + i) * ldo + gc] = v;
      }
    }
  }
}

// ---------------------------------------------------------------------------
extern "C" void kernel_launch(void* const* d_in, const int* in_sizes, int n_in,
                              void* d_out, int out_size, void* d_ws, size_t ws_size,
                              hipStream_t stream) {
  const float* x   = (const float*)d_in[0];
  const float* W1  = (const float*)d_in[1];
  const float* b1  = (const float*)d_in[2];
  const float* Wc1 = (const float*)d_in[3];
  const float* bc1 = (const float*)d_in[4];
  const float* W2  = (const float*)d_in[5];
  const float* b2  = (const float*)d_in[6];
  const float* Wc2 = (const float*)d_in[7];
  const float* bc2 = (const float*)d_in[8];
  const float* MW0 = (const float*)d_in[9];
  const float* MW1 = (const float*)d_in[10];

  char* ws = (char*)d_ws;
  unsigned short* w1bf = (unsigned short*)ws;  ws += (size_t)Z1 * K1PAD * 2;       // 1.7 MB
  float*          bias1 = (float*)ws;          ws += (size_t)Z1 * 4;               // 4 KB
  unsigned short* w2bf = (unsigned short*)ws;  ws += (size_t)OUT_CH * Z1 * 2;      // 1 MB
  float*          bias2 = (float*)ws;          ws += (size_t)OUT_CH * 4;           // 2 KB
  unsigned short* h1   = (unsigned short*)ws;  /* BATCH*Z1 bf16 = 67 MB */

  // Fold masks into bf16 weights and all bias terms into one vector per layer.
  prep_weights<<<Z1, 256, 0, stream>>>(W1, Wc1, MW0, b1, bc1, IN_CH, K1PAD, w1bf, bias1);
  prep_weights<<<OUT_CH, 256, 0, stream>>>(W2, Wc2, MW1, b2, bc2, Z1, Z1, w2bf, bias2);

  // Layer 1: h1 = relu(x * mW1^T + bias1); x converted fp32->bf16 on the fly,
  //          h1 stored bf16 (halves inter-layer HBM traffic).
  gemm_bias_relu<true, true><<<dim3(Z1 / 128, BATCH / 128), 256, 0, stream>>>(
      x, w1bf, bias1, h1, BATCH, Z1, IN_CH, IN_CH, K1PAD, Z1);

  // Layer 2: out = relu(h1 * mW2^T + bias2), fp32 output.
  gemm_bias_relu<false, false><<<dim3(OUT_CH / 128, BATCH / 128), 256, 0, stream>>>(
      h1, w2bf, bias2, d_out, BATCH, OUT_CH, Z1, Z1, Z1, OUT_CH);
}
